// Predictor_75917841924342
// MI455X (gfx1250) — matrix operand order repaired and checked by reading, not compiled, not run on hardware
//
#include <hip/hip_runtime.h>

#define SELECT_NUM 15
#define CLASS_NUM 4
#define NROI 2000
#define CROI 2048
#define FEAT 2304

typedef __attribute__((ext_vector_type(16))) __bf16 v16bf;
typedef __attribute__((ext_vector_type(8)))  float  v8f;

struct bfraw { uint4 lo; uint4 hi; };   // 32 bytes == v16bf

__device__ __forceinline__ unsigned short f2bf(float f) {
    unsigned u = __float_as_uint(f);
    u += 0x7FFFu + ((u >> 16) & 1u);     // round-to-nearest-even
    return (unsigned short)(u >> 16);
}

#define HAVE_ASYNC_LDS __has_builtin(__builtin_amdgcn_global_load_async_to_lds_b128)

#if HAVE_ASYNC_LDS
typedef int v4i __attribute__((vector_size(16)));

__device__ __forceinline__ void async_cp16(const void* g, void* l) {
    // builtin expects (AS1 v4i* src, AS3 v4i* dst, imm offset, imm cpol)
    __attribute__((address_space(1))) v4i* gp =
        (__attribute__((address_space(1))) v4i*)(unsigned long long)g;
    // generic LDS flat address: low 32 bits are the LDS offset (aperture rule)
    __attribute__((address_space(3))) v4i* lp =
        (__attribute__((address_space(3))) v4i*)(unsigned)(unsigned long long)l;
    __builtin_amdgcn_global_load_async_to_lds_b128(gp, lp, 0, 0);
}
__device__ __forceinline__ void wait_async0() {
#if __has_builtin(__builtin_amdgcn_s_wait_asynccnt)
    __builtin_amdgcn_s_wait_asynccnt(0);
#else
    asm volatile("s_wait_asynccnt 0x0" ::: "memory");
#endif
}
#endif

// ---------------- 1) ROI mean pool -> bf16 features [2000,2048] ----------------
#if HAVE_ASYNC_LDS
// Async path: each block streams one contiguous 50,176B region (256 channels x 49 f32)
// into LDS with GLOBAL_LOAD_ASYNC_TO_LDS_B128 (coalesced 512B per wave-instruction),
// then reduces from LDS (stride 49 dwords -> bank-conflict-free).
__global__ void __launch_bounds__(256) pool_roi_k(const float* __restrict__ roi,
                                                  unsigned short* __restrict__ objb) {
    __shared__ __align__(16) float xs[256 * 49];      // 50,176 B
    int bid = blockIdx.x;                              // 16000 blocks
    int n = bid >> 3, chunk = bid & 7;
    int c0 = chunk * 256;
    const char* gbase = (const char*)(roi + (long)(n * CROI + c0) * 49);  // 16B aligned
    char* lbase = (char*)xs;
    // 256ch*49 f32 = 3136 x 16B chunks
    for (int i = threadIdx.x; i < 3136; i += 256) {
        async_cp16(gbase + (long)i * 16, lbase + i * 16);
    }
    wait_async0();
    __syncthreads();
    const float* p = xs + threadIdx.x * 49;
    float s = 0.f;
    #pragma unroll
    for (int i = 0; i < 49; ++i) s += p[i];
    objb[(long)n * CROI + c0 + threadIdx.x] = f2bf(s * (1.0f / 49.0f));
}
#else
__global__ void pool_roi_k(const float* __restrict__ roi, unsigned short* __restrict__ objb) {
    int bid = blockIdx.x;                 // 16000 blocks
    int n = bid >> 3, chunk = bid & 7;
    int c = chunk * 256 + threadIdx.x;
    const float* p = roi + (long)(n * CROI + c) * 49;
    float s = 0.f;
    #pragma unroll
    for (int i = 0; i < 49; ++i) s += p[i];
    objb[(long)n * CROI + c] = f2bf(s * (1.0f / 49.0f));
}
#endif

// ---------------- 2) glob 14x14 -> 7x7 2x2 mean pool ----------------
__global__ void pool_glob_k(const float* __restrict__ g, float* __restrict__ g7) {
    int c = blockIdx.x;                   // 1024
    int t = threadIdx.x;                  // 64, use 49
    if (t >= 49) return;
    int y = t / 7, x = t % 7;
    const float* p = g + c * 196 + (2 * y) * 14 + 2 * x;
    g7[c * 49 + t] = 0.25f * (p[0] + p[1] + p[14] + p[15]);
}

// ---------------- 3) conv1 1024->512 k4 on 7x7 -> [512,4,4], relu ----------------
__global__ void conv1_k(const float* __restrict__ g7, const float* __restrict__ w,
                        const float* __restrict__ b, float* __restrict__ out) {
    int gw   = (blockIdx.x * blockDim.x + threadIdx.x) >> 5;  // 8192 waves
    int lane = threadIdx.x & 31;
    int oc = gw >> 4, pos = gw & 15, oy = pos >> 2, ox = pos & 3;
    float sum = 0.f;
    for (int ic = lane; ic < 1024; ic += 32) {
        const float* gp = g7 + ic * 49 + oy * 7 + ox;
        const float* wp = w + (oc * 1024 + ic) * 16;
        #pragma unroll
        for (int ky = 0; ky < 4; ++ky)
            #pragma unroll
            for (int kx = 0; kx < 4; ++kx)
                sum = fmaf(gp[ky * 7 + kx], wp[ky * 4 + kx], sum);
    }
    #pragma unroll
    for (int off = 16; off; off >>= 1) sum += __shfl_xor(sum, off, 32);
    if (lane == 0) out[oc * 16 + pos] = fmaxf(sum + b[oc], 0.f);
}

// ---------------- 4) conv2 512->256 k4 on 4x4 -> glob_vec[256], relu ----------------
__global__ void conv2_k(const float* __restrict__ c1, const float* __restrict__ w,
                        const float* __restrict__ b, float* __restrict__ gv) {
    int gw   = (blockIdx.x * blockDim.x + threadIdx.x) >> 5;  // 256 waves
    int lane = threadIdx.x & 31;
    int oc = gw;
    float sum = 0.f;
    for (int ic = lane; ic < 512; ic += 32) {
        const float* gp = c1 + ic * 16;
        const float* wp = w + (oc * 512 + ic) * 16;
        #pragma unroll
        for (int p = 0; p < 16; ++p) sum = fmaf(gp[p], wp[p], sum);
    }
    #pragma unroll
    for (int off = 16; off; off >>= 1) sum += __shfl_xor(sum, off, 32);
    if (lane == 0) gv[oc] = fmaxf(sum + b[oc], 0.f);
}

// ---------------- 5) convert w_sel1[:, :2048] -> bf16 ----------------
__global__ void cvt_wsel1_k(const float* __restrict__ ws, unsigned short* __restrict__ wb) {
    int i = blockIdx.x * 256 + threadIdx.x;   // 262144
    int n = i >> 11, k = i & 2047;
    wb[i] = f2bf(ws[n * FEAT + k]);
}

// ---------------- 6) selbias[n] = b_sel1[n] + w_sel1[n,2048:]·glob_vec ----------------
__global__ void selbias_k(const float* __restrict__ ws, const float* __restrict__ bs,
                          const float* __restrict__ gv, float* __restrict__ sb) {
    int gw   = (blockIdx.x * blockDim.x + threadIdx.x) >> 5;  // 128 waves
    int lane = threadIdx.x & 31;
    float sum = 0.f;
    for (int k = lane; k < 256; k += 32)
        sum = fmaf(ws[gw * FEAT + 2048 + k], gv[k], sum);
    #pragma unroll
    for (int off = 16; off; off >>= 1) sum += __shfl_xor(sum, off, 32);
    if (lane == 0) sb[gw] = sum + bs[gw];
}

// ---------------- 7) WMMA selector GEMM + fused relu·w_sel2 reduction -> scores ----------------
__global__ void __launch_bounds__(256) sel_gemm_k(
        const unsigned short* __restrict__ objb, const unsigned short* __restrict__ wb,
        const float* __restrict__ selbias, const float* __restrict__ wsel2,
        const float* __restrict__ bsel2, float* __restrict__ scores) {
    __shared__ float ssc[16];
    int tid = threadIdx.x;
    int wave = tid >> 5, lane = tid & 31;
    int half = lane >> 4, lid = lane & 15;
    int m0 = blockIdx.x * 16;             // 125 blocks
    int n0 = wave * 16;                   // 8 waves cover N=128
    if (tid < 16) ssc[tid] = 0.f;
    __syncthreads();

    // A: 16x32 bf16, lane=M, halves pick K groups (ISA 7.12.2)
    const unsigned short* arow = objb + (long)(m0 + lid) * CROI + half * 8;
    // B: 32x16 bf16, lane=N, halves pick K groups (same as A)
    const unsigned short* brow = wb + (long)(n0 + lid) * CROI + half * 8;

    v8f acc = {};
    for (int k0 = 0; k0 < CROI; k0 += 32) {
        bfraw ar, br;
        ar.lo = *(const uint4*)(arow + k0);        // K = k0 + half*8 + 0..7
        ar.hi = *(const uint4*)(arow + k0 + 16);   // K = k0 + 16 + half*8 + 0..7
        br.lo = *(const uint4*)(brow + k0);        // K = k0 + half*8 + 0..7
        br.hi = *(const uint4*)(brow + k0 + 16);   // K = k0 + 16 + half*8 + 0..7
        v16bf a = __builtin_bit_cast(v16bf, ar);
        v16bf bm = __builtin_bit_cast(v16bf, br);
        acc = __builtin_amdgcn_wmma_f32_16x16x32_bf16(
            false, a, false, bm, (short)0, acc, false, false);
    }
    // C layout: lane -> N = lid, M = r + 8*half
    int n = n0 + lid;
    float w2 = wsel2[n];
    float sb = selbias[n];
    #pragma unroll
    for (int r = 0; r < 8; ++r) {
        float v = fmaxf(acc[r] + sb, 0.f);
        atomicAdd(&ssc[r + 8 * half], v * w2);
    }
    __syncthreads();
    if (tid < 16) scores[m0 + tid] = ssc[tid] + bsel2[0];
}

// ---------------- 8) top-15 (stable) + softmax weights ----------------
__global__ void topk_k(const float* __restrict__ scores, int* __restrict__ tidx,
                       float* __restrict__ tw) {
    __shared__ float sarr[NROI];
    __shared__ float rv[256];
    __shared__ int   ri[256];
    int t = threadIdx.x;
    for (int i = t; i < NROI; i += 256) sarr[i] = scores[i];
    __syncthreads();
    // global max
    float lm = -3.0e38f;
    for (int i = t; i < NROI; i += 256) lm = fmaxf(lm, sarr[i]);
    rv[t] = lm; __syncthreads();
    for (int s = 128; s; s >>= 1) { if (t < s) rv[t] = fmaxf(rv[t], rv[t + s]); __syncthreads(); }
    float smax = rv[0]; __syncthreads();
    // sum exp
    float ls = 0.f;
    for (int i = t; i < NROI; i += 256) ls += expf(sarr[i] - smax);
    rv[t] = ls; __syncthreads();
    for (int s = 128; s; s >>= 1) { if (t < s) rv[t] += rv[t + s]; __syncthreads(); }
    float inv = 1.0f / rv[0]; __syncthreads();
    // 15 argmax rounds, tie -> lowest index (matches stable argsort of -scores)
    for (int k = 0; k < SELECT_NUM; ++k) {
        float bm = -3.0e38f; int bi = NROI;
        for (int i = t; i < NROI; i += 256)
            if (sarr[i] > bm) { bm = sarr[i]; bi = i; }
        rv[t] = bm; ri[t] = bi; __syncthreads();
        for (int s = 128; s; s >>= 1) {
            if (t < s) {
                if (rv[t + s] > rv[t] || (rv[t + s] == rv[t] && ri[t + s] < ri[t])) {
                    rv[t] = rv[t + s]; ri[t] = ri[t + s];
                }
            }
            __syncthreads();
        }
        if (t == 0) {
            tidx[k] = ri[0];
            tw[k]   = expf(rv[0] - smax) * inv;
            sarr[ri[0]] = -3.0e38f;
        }
        __syncthreads();
    }
}

// ---------------- 9) fc1: re-pool selected ROI row into LDS (exact fp32), dot ----------------
__global__ void fc1_k(const float* __restrict__ roi, const float* __restrict__ gv,
                      const int* __restrict__ tidx, const float* __restrict__ tw,
                      const float* __restrict__ w, const float* __restrict__ b,
                      float* __restrict__ z) {
    __shared__ float xs[FEAT];
    int k = blockIdx.x;                   // 15 blocks
    int j = threadIdx.x;                  // 256
    int idx = tidx[k];
    for (int c = j; c < CROI; c += 256) {
        const float* p = roi + (long)(idx * CROI + c) * 49;
        float s = 0.f;
        #pragma unroll
        for (int i = 0; i < 49; ++i) s += p[i];
        xs[c] = s * (1.0f / 49.0f);
    }
    xs[CROI + j] = gv[j];
    __syncthreads();
    const float* wr = w + j * FEAT;
    float acc = 0.f;
    for (int c = 0; c < FEAT; ++c) acc = fmaf(xs[c], wr[c], acc);
    z[k * 256 + j] = fmaxf(fmaf(tw[k], acc, b[j]), 0.f);
}

// ---------------- 10) fc2: [3840] -> [4] ----------------
__global__ void fc2_k(const float* __restrict__ z, const float* __restrict__ w,
                      const float* __restrict__ b, float* __restrict__ out) {
    int wave = threadIdx.x >> 5, lane = threadIdx.x & 31;  // 4 waves
    float s = 0.f;
    for (int i = lane; i < 256 * SELECT_NUM; i += 32)
        s = fmaf(z[i], w[wave * 256 * SELECT_NUM + i], s);
    #pragma unroll
    for (int off = 16; off; off >>= 1) s += __shfl_xor(s, off, 32);
    if (lane == 0) out[wave] = s + b[wave];
}

extern "C" void kernel_launch(void* const* d_in, const int* in_sizes, int n_in,
                              void* d_out, int out_size, void* d_ws, size_t ws_size,
                              hipStream_t stream) {
    const float* glob    = (const float*)d_in[0];
    const float* roi     = (const float*)d_in[1];
    const float* w_conv1 = (const float*)d_in[2];
    const float* b_conv1 = (const float*)d_in[3];
    const float* w_conv2 = (const float*)d_in[4];
    const float* b_conv2 = (const float*)d_in[5];
    const float* w_sel1  = (const float*)d_in[6];
    const float* b_sel1  = (const float*)d_in[7];
    const float* w_sel2  = (const float*)d_in[8];
    const float* b_sel2  = (const float*)d_in[9];
    const float* w_fc1   = (const float*)d_in[10];
    const float* b_fc1   = (const float*)d_in[11];
    const float* w_fc2   = (const float*)d_in[12];
    const float* b_fc2   = (const float*)d_in[13];
    float* out = (float*)d_out;

    // workspace layout (all 16B aligned), ~9 MB total
    char* ws = (char*)d_ws;
    size_t off = 0;
    unsigned short* objb   = (unsigned short*)(ws + off); off += (size_t)NROI * CROI * 2;     // 8,192,000
    unsigned short* wselb  = (unsigned short*)(ws + off); off += (size_t)128 * CROI * 2;      //   524,288
    float* g7      = (float*)(ws + off); off += 1024 * 49 * 4;                                //   200,704
    float* c1out   = (float*)(ws + off); off += 512 * 16 * 4;
    float* glob_v  = (float*)(ws + off); off += 256 * 4;
    float* selbias = (float*)(ws + off); off += 128 * 4;
    float* scores  = (float*)(ws + off); off += NROI * 4;
    int*   tidx    = (int*)(ws + off);   off += 16 * 4;
    float* tw      = (float*)(ws + off); off += 16 * 4;
    float* z       = (float*)(ws + off); off += SELECT_NUM * 256 * 4;

    pool_roi_k <<<NROI * 8, 256, 0, stream>>>(roi, objb);
    pool_glob_k<<<1024, 64, 0, stream>>>(glob, g7);
    conv1_k    <<<1024, 256, 0, stream>>>(g7, w_conv1, b_conv1, c1out);
    conv2_k    <<<32, 256, 0, stream>>>(c1out, w_conv2, b_conv2, glob_v);
    cvt_wsel1_k<<<1024, 256, 0, stream>>>(w_sel1, wselb);
    selbias_k  <<<16, 256, 0, stream>>>(w_sel1, b_sel1, glob_v, selbias);
    sel_gemm_k <<<NROI / 16, 256, 0, stream>>>(objb, wselb, selbias, w_sel2, b_sel2, scores);
    topk_k     <<<1, 256, 0, stream>>>(scores, tidx, tw);
    fc1_k      <<<SELECT_NUM, 256, 0, stream>>>(roi, glob_v, tidx, tw, w_fc1, b_fc1, z);
    fc2_k      <<<1, 128, 0, stream>>>(z, w_fc2, b_fc2, out);
}
